// LinearAtten_163208757401
// MI455X (gfx1250) — compile-verified, matrix operand order checked by reading, not run on hardware
//
#include <hip/hip_runtime.h>

typedef __attribute__((ext_vector_type(16))) __bf16 v16bf;
typedef __attribute__((ext_vector_type(8)))  __bf16 v8bf;
typedef __attribute__((ext_vector_type(4)))  __bf16 v4bf;
typedef __attribute__((ext_vector_type(8)))  float  v8f;
typedef __attribute__((ext_vector_type(4)))  float  v4f;

#define LTOT 16384
#define CCH  128
#define NH   4
#define DH   32

__device__ __forceinline__ v8f wmma_bf16(v16bf a, v16bf b, v8f c) {
  // D = A(16x32 bf16) x B(32x16 bf16) + C(16x16 f32)
  return __builtin_amdgcn_wmma_f32_16x16x32_bf16(false, a, false, b, (short)0, c,
                                                 false, false);
}

// Build a 16x32 bf16 A/B fragment from row-major fp32 memory.
// CDNA5 16-bit A layout: lane L holds row (L&15); halves 0..7 are K = k0+{0..7}
// (lanes<16) or k0+{8..15} (lanes>=16); halves 8..15 are +16 beyond that.
__device__ __forceinline__ v16bf load_frag_f32(const float* __restrict__ base, int ld,
                                               int row0, int k0, int lane) {
  const float* p = base + (size_t)(row0 + (lane & 15)) * ld + k0 + ((lane & 16) ? 8 : 0);
  v4f f0 = *(const v4f*)(p + 0);
  v4f f1 = *(const v4f*)(p + 4);
  v4f f2 = *(const v4f*)(p + 16);
  v4f f3 = *(const v4f*)(p + 20);
  v16bf o;
#pragma unroll
  for (int i = 0; i < 4; ++i) {
    o[i]      = (__bf16)f0[i];
    o[4 + i]  = (__bf16)f1[i];
    o[8 + i]  = (__bf16)f2[i];
    o[12 + i] = (__bf16)f3[i];
  }
  return o;
}

// Same fragment from bf16 LDS (row stride ldh halves; row*ldh*2 and k0*2 16B aligned)
__device__ __forceinline__ v16bf load_frag_lds(const __bf16* base, int ldh,
                                               int row0, int k0, int lane) {
  const __bf16* p = base + (row0 + (lane & 15)) * ldh + k0 + ((lane & 16) ? 8 : 0);
  v8bf a = *(const v8bf*)(p);
  v8bf b = *(const v8bf*)(p + 16);
  v16bf o;
#pragma unroll
  for (int i = 0; i < 8; ++i) { o[i] = a[i]; o[8 + i] = b[i]; }
  return o;
}

__device__ __forceinline__ void cvt_store4(__bf16* d, v4f f) {
  v4bf o;
  o[0] = (__bf16)f[0]; o[1] = (__bf16)f[1]; o[2] = (__bf16)f[2]; o[3] = (__bf16)f[3];
  *(v4bf*)d = o;   // 8B-aligned -> ds_write_b64
}

// ---------------- Kernel A: k/v projection + exp(k), reduce over L into S,Z --------
// grid 512 = 8 batches * 64 chunks of 256 positions; 256 threads (8 waves)
__global__ __launch_bounds__(256) void lat_kv_reduce(const float* __restrict__ x,
                                                     const float* __restrict__ Wqkv,
                                                     float* __restrict__ S,
                                                     float* __restrict__ Z) {
  __shared__ __bf16 xs[32 * 136];  // staged x: [pos 0..31][chan], stride 136
  __shared__ __bf16 ek[128 * 40];  // exp(k): [kchan 0..127][chunk col 0..31], stride 40
  __shared__ __bf16 vt[128 * 40];  // v     : [vchan 0..127][chunk col 0..31]
  const int b    = blockIdx.x >> 6;
  const int l0   = (blockIdx.x & 63) * 256;
  const int tid  = threadIdx.x;
  const int w    = tid >> 5;
  const int lane = tid & 31;

  // W rows for k (global 128..255) and v (256..383); wave w owns rows 32w..32w+31
  const float* Wkv = Wqkv + 128 * CCH;
  v16bf afr[2][4];
#pragma unroll
  for (int t = 0; t < 2; ++t)
#pragma unroll
    for (int ks = 0; ks < 4; ++ks)
      afr[t][ks] = load_frag_f32(Wkv, CCH, 32 * w + 16 * t, 32 * ks, lane);

  v16bf ones;
#pragma unroll
  for (int i = 0; i < 16; ++i) ones[i] = (__bf16)1.0f;

  const int h  = w >> 1;   // head this wave reduces
  const int rt = w & 1;    // row-tile (c 0..15 / 16..31) within head
  v8f accS0 = {}, accS1 = {}, accZ = {};

  const float* xb = x + ((size_t)b * LTOT + l0) * CCH;
  const bool isK = (w < 4);
  __bf16* dst = isK ? ek : vt;
  const int rsec = 32 * w - (isK ? 0 : 128);

  for (int ch = 0; ch < 8; ++ch) {          // 8 sub-chunks of 32 positions
    // stage 32 positions (32x128 fp32 -> bf16), fully coalesced
    {
      const float* src = xb + (size_t)(ch * 32) * CCH;
#pragma unroll
      for (int i = 0; i < 4; ++i) {
        const int idx = tid + 256 * i;       // 1024 float4s
        const int pos = idx >> 5;
        const int cc  = (idx & 31) * 4;
        v4f f = *(const v4f*)(src + pos * CCH + cc);
        cvt_store4(&xs[pos * 136 + cc], f);
        if (ch < 7) __builtin_prefetch(src + 32 * CCH + pos * CCH + cc, 0, 1);
      }
    }
    __syncthreads();
    // GEMM1: rows 32w..32w+31 of [k;v], cols of this chunk
#pragma unroll
    for (int t = 0; t < 2; ++t) {
#pragma unroll
      for (int lt = 0; lt < 2; ++lt) {
        v8f acc = {};
#pragma unroll
        for (int ks = 0; ks < 4; ++ks) {
          v16bf bfr = load_frag_lds(xs, 136, 16 * lt, 32 * ks, lane);
          acc = wmma_bf16(afr[t][ks], bfr, acc);
        }
        const int coll = 16 * lt + (lane & 15);
#pragma unroll
        for (int r = 0; r < 8; ++r) {
          const int rr = rsec + 16 * t + r + ((lane & 16) ? 8 : 0);
          float v = acc[r];
          if (isK) v = __expf(v);
          dst[rr * 40 + coll] = (__bf16)v;
        }
      }
    }
    __syncthreads();
    // GEMM2: S_h[c,d] += exp(k)_h[c,:] * v_h[d,:]^T over this 32-wide K slab
    v16bf a  = load_frag_lds(ek, 40, 32 * h + 16 * rt, 0, lane);
    v16bf b0 = load_frag_lds(vt, 40, 32 * h + 0,       0, lane);
    v16bf b1 = load_frag_lds(vt, 40, 32 * h + 16,      0, lane);
    accS0 = wmma_bf16(a, b0, accS0);
    accS1 = wmma_bf16(a, b1, accS1);
    accZ  = wmma_bf16(a, ones, accZ);  // every column of D equals Z_c
    __syncthreads();
  }

#pragma unroll
  for (int r = 0; r < 8; ++r) {
    const int c  = 16 * rt + r + ((lane & 16) ? 8 : 0);
    const int d0 = lane & 15;
    float* Sp = S + (((b * NH + h) * DH + c) * DH);
    atomicAdd(Sp + d0,      accS0[r]);
    atomicAdd(Sp + 16 + d0, accS1[r]);
  }
  if ((lane & 15) == 0) {
#pragma unroll
    for (int r = 0; r < 8; ++r) {
      const int c = 16 * rt + r + ((lane & 16) ? 8 : 0);
      atomicAdd(&Z[(b * NH + h) * DH + c], accZ[r]);
    }
  }
}

// ------- Kernel C: fused q-proj + head-dim softmax + ctx*q + W_out + GN partials ---
// grid 1024 = 8 batches * 128 tiles of 128 positions; 256 threads (8 waves)
__global__ __launch_bounds__(256) void lat_fused_out(const float* __restrict__ x,
                                                     const float* __restrict__ Wqkv,
                                                     const float* __restrict__ Wout,
                                                     const float* __restrict__ bout,
                                                     const float* __restrict__ S,
                                                     const float* __restrict__ Z,
                                                     float* __restrict__ Y,
                                                     float* __restrict__ gsum,
                                                     float* __restrict__ gsq) {
  __shared__ __bf16 qls[128 * 136];  // [pos][chan], stride 136; reused for attn
  __shared__ __bf16 ctx[128 * 40];   // [chan(h*32+c)][d], stride 40
  __shared__ __bf16 xs[64 * 136];    // staged x half-tile
  __shared__ float  red[2];
  const int b    = blockIdx.x >> 7;
  const int l0   = (blockIdx.x & 127) * 128;
  const int tid  = threadIdx.x;
  const int w    = tid >> 5;
  const int lane = tid & 31;
  if (tid == 0) { red[0] = 0.0f; red[1] = 0.0f; }

  // context = S / Z in bf16 LDS
  if (tid < 128) {
    const int hh = tid >> 5, c = tid & 31;
    const float inv = 1.0f / Z[(b * NH + hh) * DH + c];
    const float* Sp = S + (((b * NH + hh) * DH + c) * DH);
#pragma unroll
    for (int d = 0; d < 32; ++d) ctx[tid * 40 + d] = (__bf16)(Sp[d] * inv);
  }

  // GEMM q: wave w -> q channels 16w..16w+15 (W_qkv rows 0..127)
  v16bf aq[4];
#pragma unroll
  for (int ks = 0; ks < 4; ++ks)
    aq[ks] = load_frag_f32(Wqkv, CCH, 16 * w, 32 * ks, lane);
  const float* xb = x + ((size_t)b * LTOT + l0) * CCH;

  for (int half = 0; half < 2; ++half) {
    // stage 64 positions of x as bf16
    const float* src = xb + (size_t)(half * 64) * CCH;
#pragma unroll
    for (int i = 0; i < 8; ++i) {
      const int idx = tid + 256 * i;       // 2048 float4s
      const int pos = idx >> 5;
      const int cc  = (idx & 31) * 4;
      v4f f = *(const v4f*)(src + pos * CCH + cc);
      cvt_store4(&xs[pos * 136 + cc], f);
      if (half == 0) __builtin_prefetch(src + 64 * CCH + pos * CCH + cc, 0, 1);
    }
    __syncthreads();
#pragma unroll
    for (int lt = 0; lt < 4; ++lt) {
      v8f acc = {};
#pragma unroll
      for (int ks = 0; ks < 4; ++ks) {
        v16bf bq = load_frag_lds(xs, 136, 16 * lt, 32 * ks, lane);
        acc = wmma_bf16(aq[ks], bq, acc);
      }
      const int pos = half * 64 + 16 * lt + (lane & 15);
#pragma unroll
      for (int r = 0; r < 8; ++r)
        qls[pos * 136 + 16 * w + r + ((lane & 16) ? 8 : 0)] = (__bf16)acc[r];
    }
    __syncthreads();
  }

  // per-position softmax over head-dim (32), times d^-0.5, in place
  const float qscale = 0.17677669529663687f;  // 32^-0.5
  for (int task = tid; task < 512; task += 256) {
    const int pos = task >> 2, hh = task & 3;
    __bf16* p = &qls[pos * 136 + 32 * hh];
    float vals[32];
    float m = -1e30f;
#pragma unroll
    for (int i = 0; i < 32; ++i) { vals[i] = (float)p[i]; m = fmaxf(m, vals[i]); }
    float s = 0.0f;
#pragma unroll
    for (int i = 0; i < 32; ++i) { vals[i] = __expf(vals[i] - m); s += vals[i]; }
    const float inv = qscale / s;
#pragma unroll
    for (int i = 0; i < 32; ++i) p[i] = (__bf16)(vals[i] * inv);
  }
  __syncthreads();

  // GEMM attn: out chan rows 16w..16w+15 (head h=w/2), K = 32 head dims
  {
    const int hh = w >> 1;
    v16bf a = load_frag_lds(ctx, 40, 16 * w, 0, lane);
    v8f acc[8];
#pragma unroll
    for (int lt = 0; lt < 8; ++lt) {
      v16bf bfr = load_frag_lds(qls, 136, 16 * lt, 32 * hh, lane);
      v8f zz = {};
      acc[lt] = wmma_bf16(a, bfr, zz);
    }
    __syncthreads();  // everyone done reading qls
#pragma unroll
    for (int lt = 0; lt < 8; ++lt) {
      const int pos = 16 * lt + (lane & 15);
#pragma unroll
      for (int r = 0; r < 8; ++r)
        qls[pos * 136 + 16 * w + r + ((lane & 16) ? 8 : 0)] = (__bf16)acc[lt][r];
    }
  }
  __syncthreads();

  // GEMM y = W_out * attn + b_out; accumulate GroupNorm partials; store Y[b][c][l]
  v16bf ao[4];
#pragma unroll
  for (int ks = 0; ks < 4; ++ks)
    ao[ks] = load_frag_f32(Wout, CCH, 16 * w, 32 * ks, lane);
  float lsum = 0.0f, lsq = 0.0f;
#pragma unroll
  for (int lt = 0; lt < 8; ++lt) {
    v8f acc = {};
#pragma unroll
    for (int ks = 0; ks < 4; ++ks) {
      v16bf bf = load_frag_lds(qls, 136, 16 * lt, 32 * ks, lane);
      acc = wmma_bf16(ao[ks], bf, acc);
    }
    const int pos = l0 + 16 * lt + (lane & 15);
#pragma unroll
    for (int r = 0; r < 8; ++r) {
      const int row = 16 * w + r + ((lane & 16) ? 8 : 0);
      const float yv = acc[r] + bout[row];
      Y[((size_t)(b * CCH + row)) * LTOT + pos] = yv;
      lsum += yv;
      lsq  += yv * yv;
    }
  }
  atomicAdd(&red[0], lsum);
  atomicAdd(&red[1], lsq);
  __syncthreads();
  if (tid == 0) { atomicAdd(&gsum[b], red[0]); atomicAdd(&gsq[b], red[1]); }
}

// ---------------- Kernel D: GroupNorm finalize + [b,c,l] -> [b,l,c] transpose ------
__global__ __launch_bounds__(256) void lat_gn_transpose(const float* __restrict__ Y,
                                                        const float* __restrict__ gsum,
                                                        const float* __restrict__ gsq,
                                                        const float* __restrict__ gw,
                                                        const float* __restrict__ gb,
                                                        float* __restrict__ out) {
  __shared__ float tile[128][65];
  const int b  = blockIdx.x >> 8;   // 256 l-chunks of 64
  const int l0 = (blockIdx.x & 255) * 64;
  const float n    = (float)CCH * (float)LTOT;
  const float mean = gsum[b] / n;
  const float var  = gsq[b] / n - mean * mean;
  const float rinv = rsqrtf(var + 1e-5f);
  const int tid = threadIdx.x;
#pragma unroll
  for (int i = 0; i < 32; ++i) {
    const int e = tid + 256 * i;
    const int c = e >> 6, l = e & 63;
    tile[c][l] = Y[((size_t)(b * CCH + c)) * LTOT + l0 + l];
  }
  __syncthreads();
#pragma unroll
  for (int i = 0; i < 32; ++i) {
    const int e = tid + 256 * i;
    const int l = e >> 7, c = e & 127;
    const float v = (tile[c][l] - mean) * rinv * gw[c] + gb[c];
    out[((size_t)b * LTOT + l0 + l) * CCH + c] = v;
  }
}

extern "C" void kernel_launch(void* const* d_in, const int* in_sizes, int n_in,
                              void* d_out, int out_size, void* d_ws, size_t ws_size,
                              hipStream_t stream) {
  (void)in_sizes; (void)n_in; (void)out_size; (void)ws_size;
  const float* x    = (const float*)d_in[0];
  const float* Wqkv = (const float*)d_in[1];
  const float* Wout = (const float*)d_in[2];
  const float* bout = (const float*)d_in[3];
  const float* gw   = (const float*)d_in[4];
  const float* gb   = (const float*)d_in[5];

  float* ws   = (float*)d_ws;
  float* Y    = ws;                          // 8*128*16384 = 16,777,216 floats
  float* S    = ws + (size_t)16777216;       // 8*4*32*32   = 32,768
  float* Z    = S + 32768;                   // 8*4*32      = 1,024
  float* gsum = Z + 1024;                    // 8
  float* gsq  = gsum + 8;                    // 8

  hipMemsetAsync(S, 0, (32768 + 1024 + 16) * sizeof(float), stream);
  lat_kv_reduce  <<<512,  256, 0, stream>>>(x, Wqkv, S, Z);
  lat_fused_out  <<<1024, 256, 0, stream>>>(x, Wqkv, Wout, bout, S, Z, Y, gsum, gsq);
  lat_gn_transpose<<<2048, 256, 0, stream>>>(Y, gsum, gsq, gw, gb, (float*)d_out);
}
